// SSGC_23673859736193
// MI455X (gfx1250) — compile-verified, matrix operand order checked by reading, not run on hardware
//
#include <hip/hip_runtime.h>
#include <hip/hip_bf16.h>

typedef __attribute__((ext_vector_type(16))) _Float16 v16h;
typedef __attribute__((ext_vector_type(8)))  _Float16 v8h;
typedef __attribute__((ext_vector_type(8)))  float    v8f;

#define KPROP 10
#define ALPHA 0.1f

// ---------------------------------------------------------------------------
// fill: p[i] = v
__global__ __launch_bounds__(256) void ssgc_fill(float* __restrict__ p, float v, long long n) {
    long long i = (long long)blockIdx.x * blockDim.x + threadIdx.x;
    if (i < n) p[i] = v;
}

// degree scatter: deg[row[e]] += w[e]  (deg pre-filled with 1.0 for self loops)
__global__ __launch_bounds__(256) void ssgc_deg(const int* __restrict__ row,
                                                const float* __restrict__ w,
                                                float* __restrict__ deg, int e_total) {
    int e = blockIdx.x * blockDim.x + threadIdx.x;
    if (e < e_total) atomicAdd(&deg[row[e]], w[e]);
}

// dinv[i] = rsqrt(max(deg,eps)); sw[i] = dinv*dinv (self-loop normalized weight)
__global__ __launch_bounds__(256) void ssgc_dinv(const float* __restrict__ deg,
                                                 float* __restrict__ dinv,
                                                 float* __restrict__ sw, int n) {
    int i = blockIdx.x * blockDim.x + threadIdx.x;
    if (i < n) {
        float d = deg[i];
        d = d > 1e-12f ? d : 1e-12f;
        float r = rsqrtf(d);
        dinv[i] = r;
        sw[i]   = r * r;
    }
}

// nw[e] = dinv[row]*w*dinv[col]
__global__ __launch_bounds__(256) void ssgc_nw(const int* __restrict__ row,
                                               const int* __restrict__ col,
                                               const float* __restrict__ w,
                                               const float* __restrict__ dinv,
                                               float* __restrict__ nw, int e_total) {
    int e = blockIdx.x * blockDim.x + threadIdx.x;
    if (e < e_total) nw[e] = dinv[row[e]] * w[e] * dinv[col[e]];
}

// self-loop init (+ optional fused acc accumulation of the PREVIOUS hop):
//   nxt[i][:] = sw[i]*cur[i][:];   if (addacc) acc[i][:] += cur[i][:]
__global__ __launch_bounds__(256) void ssgc_self(float* __restrict__ nxt,
                                                 const float* __restrict__ cur,
                                                 const float* __restrict__ sw,
                                                 float* __restrict__ acc,
                                                 int addacc, int n) {
    long long tid = (long long)blockIdx.x * blockDim.x + threadIdx.x;
    if (tid >= (long long)n * 16) return;
    int node = (int)(tid >> 4);
    int fq   = ((int)tid & 15) << 2;
    long long off = (long long)node * 64 + fq;
    float s = sw[node];
    const float4 h = *(const float4*)(cur + off);
    float4 o; o.x = s * h.x; o.y = s * h.y; o.z = s * h.z; o.w = s * h.w;
    *(float4*)(nxt + off) = o;
    if (addacc) {
        float4 a = *(float4*)(acc + off);
        a.x += h.x; a.y += h.y; a.z += h.z; a.w += h.w;
        *(float4*)(acc + off) = a;
    }
}

// edge scatter: nxt[col][f] += nw[e] * cur[row][f]  (16 lanes/edge, float4 gather)
__global__ __launch_bounds__(256) void ssgc_edge(const int* __restrict__ row,
                                                 const int* __restrict__ col,
                                                 const float* __restrict__ nw,
                                                 const float* __restrict__ cur,
                                                 float* __restrict__ nxt, int e_total) {
    long long tid = (long long)blockIdx.x * blockDim.x + threadIdx.x;
    int e = (int)(tid >> 4);
    if (e >= e_total) return;
    int fq = ((int)tid & 15) << 2;
    int r = row[e], c = col[e];
    float wt = nw[e];
    const float4 h = *(const float4*)(cur + (long long)r * 64 + fq);
    float* dst = nxt + (long long)c * 64 + fq;
    atomicAdd(dst + 0, wt * h.x);
    atomicAdd(dst + 1, wt * h.y);
    atomicAdd(dst + 2, wt * h.z);
    atomicAdd(dst + 3, wt * h.w);
}

// acc += h  (float4) — used once after the K-loop for the last hop
__global__ __launch_bounds__(256) void ssgc_accadd(float* __restrict__ acc,
                                                   const float* __restrict__ h, long long nq) {
    long long i = (long long)blockIdx.x * blockDim.x + threadIdx.x;
    if (i >= nq) return;
    float4 a = *(float4*)(acc + i * 4);
    const float4 b = *(const float4*)(h + i * 4);
    a.x += b.x; a.y += b.y; a.z += b.z; a.w += b.w;
    *(float4*)(acc + i * 4) = a;
}

// ---------------------------------------------------------------------------
// One-off weight pack: W0 (64x64) -> 8 B-fragments, W1 (64x32) -> 4 B-fragments,
// f16, laid out so each lane reads its 16 elements as 32 contiguous bytes.
// Fragment (nt,s): element e of lane l holds W[(s*32 + k(e,hi))*H + nt*16 + m],
// k(e,hi) = 16*(e>>3) + 8*hi + (e&7), m = l&15, hi = l>>4.
__global__ __launch_bounds__(256) void ssgc_packW(const float* __restrict__ W0,
                                                  const float* __restrict__ W1,
                                                  _Float16* __restrict__ W0f,
                                                  _Float16* __restrict__ W1f) {
    int t = blockIdx.x * blockDim.x + threadIdx.x;   // (frag, lane)
    int lane = t & 31;
    int frag = t >> 5;
    int m = lane & 15, hi = lane >> 4;
    if (frag < 8) {                                   // W0: frag = nt*2 + s
        int nt = frag >> 1, s = frag & 1;
        _Float16* dst = W0f + ((size_t)frag * 32 + lane) * 16;
#pragma unroll
        for (int e = 0; e < 16; ++e) {
            int k = ((e >> 3) << 4) + (hi << 3) + (e & 7);
            dst[e] = (_Float16)W0[(s * 32 + k) * 64 + nt * 16 + m];
        }
    } else if (frag < 12) {                           // W1: frag-8 = nt*2 + s
        int f = frag - 8;
        int nt = f >> 1, s = f & 1;
        _Float16* dst = W1f + ((size_t)f * 32 + lane) * 16;
#pragma unroll
        for (int e = 0; e < 16; ++e) {
            int k = ((e >> 3) << 4) + (hi << 3) + (e & 7);
            dst[e] = (_Float16)W1[(s * 32 + k) * 32 + nt * 16 + m];
        }
    }
}

// ---------------------------------------------------------------------------
// Fused mix + 2-layer MLP on WMMA. One wave32 per 16-row tile.
//   Y = acc*(0.9/K) + ALPHA*x        (A-frags built from 8 float4 loads/operand)
//   Z = relu(Y @ W0 + b0)            (staged row-major f16 in per-wave LDS)
//   OUT = Z @ W1 + b1
__global__ __launch_bounds__(256) void ssgc_mlp_wmma(
    const float* __restrict__ acc, const float* __restrict__ x,
    const v16h* __restrict__ W0v, const float* __restrict__ bias0,
    const v16h* __restrict__ W1v, const float* __restrict__ bias1,
    float* __restrict__ out, int n) {
    __shared__ __align__(32) _Float16 zbuf[8][16 * 64];
    const int wave = threadIdx.x >> 5;
    const int lane = threadIdx.x & 31;
    const int m    = lane & 15;
    const int hi   = lane >> 4;
    const int tile = blockIdx.x * 8 + wave;
    const int rowBase = tile * 16;
    if (rowBase >= n) return;   // uniform per wave -> EXEC stays all-ones inside

    const float c1 = (1.0f - ALPHA) / (float)KPROP;
    _Float16* z = zbuf[wave];

    // ---- A fragments for layer 1 (two k=32 steps), mix folded in.
    // Per lane: four contiguous 8-float runs at float offsets 0,16,32,48 from
    // (row*64 + 8*hi); run j -> float4 indices (vi, vi+1).
    const float4* accr = (const float4*)(acc + (long long)(rowBase + m) * 64 + (hi << 3));
    const float4* xr   = (const float4*)(x   + (long long)(rowBase + m) * 64 + (hi << 3));
    float ya[32];
#pragma unroll
    for (int j = 0; j < 4; ++j) {
        int vi = j * 4;                       // float4 index: runs at floats 0,16,32,48
        float4 pa = accr[vi], pb = accr[vi + 1];
        float4 qa = xr[vi],   qb = xr[vi + 1];
        ya[j * 8 + 0] = pa.x * c1 + qa.x * ALPHA;
        ya[j * 8 + 1] = pa.y * c1 + qa.y * ALPHA;
        ya[j * 8 + 2] = pa.z * c1 + qa.z * ALPHA;
        ya[j * 8 + 3] = pa.w * c1 + qa.w * ALPHA;
        ya[j * 8 + 4] = pb.x * c1 + qb.x * ALPHA;
        ya[j * 8 + 5] = pb.y * c1 + qb.y * ALPHA;
        ya[j * 8 + 6] = pb.z * c1 + qb.z * ALPHA;
        ya[j * 8 + 7] = pb.w * c1 + qb.w * ALPHA;
    }
    v16h a0, a1;
#pragma unroll
    for (int e = 0; e < 16; ++e) {
        a0[e] = (_Float16)ya[e];
        a1[e] = (_Float16)ya[16 + e];
    }

    // ---- Layer 1: 4 column tiles of W0, B-fragments as packed v16h loads
#pragma unroll
    for (int nt = 0; nt < 4; ++nt) {
        int ncol = nt * 16 + m;
        v16h bf0 = W0v[(nt * 2 + 0) * 32 + lane];
        v16h bf1 = W0v[(nt * 2 + 1) * 32 + lane];
        v8f c = {};
        c = __builtin_amdgcn_wmma_f32_16x16x32_f16(false, a0, false, bf0, (short)0, c, false, false);
        c = __builtin_amdgcn_wmma_f32_16x16x32_f16(false, a1, false, bf1, (short)0, c, false, false);
        float bv = bias0[ncol];
#pragma unroll
        for (int r = 0; r < 8; ++r) {
            float v = c[r] + bv;
            v = v > 0.0f ? v : 0.0f;                       // relu
            z[(r + (hi << 3)) * 64 + ncol] = (_Float16)v;  // D layout: row=r+8*hi, col=m
        }
    }

    // single-wave tile: only need DS ordering, not a workgroup barrier
    asm volatile("s_wait_dscnt 0" ::: "memory");

    // ---- A fragments for layer 2: 4 contiguous v8h (ds_load_b128) + shuffles
    const v8h* zr = (const v8h*)(z + m * 64 + (hi << 3));
    v8h z0 = zr[0], z1 = zr[2], z2 = zr[4], z3 = zr[6];
    v16h za0 = __builtin_shufflevector(z0, z1, 0,1,2,3,4,5,6,7,8,9,10,11,12,13,14,15);
    v16h za1 = __builtin_shufflevector(z2, z3, 0,1,2,3,4,5,6,7,8,9,10,11,12,13,14,15);

    // ---- Layer 2: 2 column tiles of W1
#pragma unroll
    for (int nt = 0; nt < 2; ++nt) {
        int ncol = nt * 16 + m;
        v16h wf0 = W1v[(nt * 2 + 0) * 32 + lane];
        v16h wf1 = W1v[(nt * 2 + 1) * 32 + lane];
        v8f c = {};
        c = __builtin_amdgcn_wmma_f32_16x16x32_f16(false, za0, false, wf0, (short)0, c, false, false);
        c = __builtin_amdgcn_wmma_f32_16x16x32_f16(false, za1, false, wf1, (short)0, c, false, false);
        float bv = bias1[ncol];
#pragma unroll
        for (int r = 0; r < 8; ++r)
            out[(long long)(rowBase + r + (hi << 3)) * 32 + ncol] = c[r] + bv;
    }
}

// ---------------------------------------------------------------------------
extern "C" void kernel_launch(void* const* d_in, const int* in_sizes, int n_in,
                              void* d_out, int out_size, void* d_ws, size_t ws_size,
                              hipStream_t stream) {
    const float* x  = (const float*)d_in[0];
    const int*   ei = (const int*)d_in[1];
    const float* ew = (const float*)d_in[2];
    const float* W0 = (const float*)d_in[3];
    const float* b0 = (const float*)d_in[4];
    const float* W1 = (const float*)d_in[5];
    const float* b1 = (const float*)d_in[6];
    float* out = (float*)d_out;

    const int N = in_sizes[0] / 64;
    const int E = in_sizes[2];
    const int* row = ei;
    const int* col = ei + E;

    // workspace layout: f16 weight fragments first (32B-aligned), then floats
    _Float16* W0f = (_Float16*)d_ws;                 // 8 frags * 32 lanes * 16 = 8KB
    _Float16* W1f = W0f + 8 * 32 * 16;               // 4 frags               = 4KB
    float* ws   = (float*)((char*)d_ws + 12 * 1024);
    float* deg  = ws;
    float* dinv = deg  + N;
    float* sw   = dinv + N;
    float* nw   = sw   + N;
    size_t base = ((size_t)3 * N + (size_t)E + 3) & ~(size_t)3;  // 16B align
    float* h0   = ws + base;
    float* h1   = h0 + (size_t)N * 64;
    float* acc  = h1 + (size_t)N * 64;

    const int B = 256;
    const long long nfeat = (long long)N * 64;
    const long long nquad = (long long)N * 16;

    // normalization + weight packing
    ssgc_fill<<<(N + B - 1) / B, B, 0, stream>>>(deg, 1.0f, N);
    ssgc_deg <<<(E + B - 1) / B, B, 0, stream>>>(row, ew, deg, E);
    ssgc_dinv<<<(N + B - 1) / B, B, 0, stream>>>(deg, dinv, sw, N);
    ssgc_nw  <<<(E + B - 1) / B, B, 0, stream>>>(row, col, ew, dinv, nw, E);
    ssgc_packW<<<2, B, 0, stream>>>(W0, W1, W0f, W1f);

    // acc = 0
    ssgc_fill<<<(int)((nfeat + B - 1) / B), B, 0, stream>>>(acc, 0.0f, nfeat);

    // K propagation steps; acc += h_k fused into the next step's self kernel
    const float* cur = x;
    float* nxt = h0;
    const int gq = (int)((nquad + B - 1) / B);
    const int ge = (int)(((long long)E * 16 + B - 1) / B);
    for (int k = 0; k < KPROP; ++k) {
        // k==0 reads x (not part of acc); k>0 folds acc += h_k while reading it
        ssgc_self<<<gq, B, 0, stream>>>(nxt, cur, sw, acc, k > 0 ? 1 : 0, N);
        ssgc_edge<<<ge, B, 0, stream>>>(row, col, nw, cur, nxt, E);
        cur = nxt;
        nxt = (nxt == h0) ? h1 : h0;
    }
    ssgc_accadd<<<gq, B, 0, stream>>>(acc, cur, nquad);  // add final hop h_K

    // fused mix + MLP (WMMA)
    int tiles  = (N + 15) / 16;
    int blocks = (tiles + 7) / 8;
    ssgc_mlp_wmma<<<blocks, 256, 0, stream>>>(acc, x, (const v16h*)W0f, b0,
                                              (const v16h*)W1f, b1, out, N);
}
